// HSTURecTaobao_38783554683550
// MI455X (gfx1250) — compile-verified, matrix operand order checked by reading, not run on hardware
//
#include <hip/hip_runtime.h>

typedef __attribute__((ext_vector_type(16))) __bf16 v16bf;
typedef __attribute__((ext_vector_type(8)))  float  v8f;

struct alignas(16) F4 { float x, y, z, w; };
struct F8 { F4 a, b; };

static __device__ __forceinline__ v16bf make_frag(const __bf16* lo, const __bf16* hi) {
  F8 t;
  t.a = *(const F4*)lo;
  t.b = *(const F4*)hi;
  return __builtin_bit_cast(v16bf, t);
}

static __device__ __forceinline__ __bf16 f2bf(float f) {
  unsigned u = __builtin_bit_cast(unsigned, f);
  unsigned r = (u + 0x7FFFu + ((u >> 16) & 1u)) >> 16;
  unsigned short s = (unsigned short)r;
  return __builtin_bit_cast(__bf16, s);
}

// pack two floats as bf16 pair (lo in [15:0], hi in [31:16]) with RNE
static __device__ __forceinline__ unsigned pack_bf16(float a, float b) {
  unsigned ua = __builtin_bit_cast(unsigned, a);
  unsigned ub = __builtin_bit_cast(unsigned, b);
  unsigned ra = (ua + 0x7FFFu + ((ua >> 16) & 1u)) >> 16;
  unsigned rb = (ub + 0x7FFFu + ((ub >> 16) & 1u)) & 0xFFFF0000u;
  return ra | rb;
}

static __device__ __forceinline__ v8f wmma_bf16(v16bf a, v16bf b, v8f c) {
  return __builtin_amdgcn_wmma_f32_16x16x32_bf16(false, a, false, b, (short)0, c, false, false);
}

// fast sigmoid / silu via v_rcp_f32 (avoids IEEE division expansion)
static __device__ __forceinline__ float fsigmoid(float x) {
  return __builtin_amdgcn_rcpf(1.f + __expf(-x));
}
static __device__ __forceinline__ float silu(float x) {
  return x * fsigmoid(x);
}

// ---------------- K0: pair table + time-bucket boundaries ----------------
__global__ void k_prep(const float* __restrict__ raw, float* __restrict__ table,
                       float* __restrict__ bounds, int* __restrict__ nbp) {
  int t = threadIdx.x;
  if (t < 200) {
    const float P[25] = {0.f,0.f,0.f,0.f,0.f,
                         0.f,0.05f,0.f,0.f,0.f,
                         0.f,0.35f,0.05f,0.f,0.f,
                         0.f,0.25f,0.85f,0.05f,0.f,
                         0.f,0.25f,0.8f,1.2f,0.1f};
    table[t] = P[t % 25] + 0.25f * tanhf(raw[t]);
  }
  if (t == 0) {
    double lmax = log(86400.0 * 90.0);
    int n = 0; double prev = -1.0;
    for (int i = 0; i < 63; ++i) {
      double v = floor(exp(lmax * (double)i / 62.0));
      if (v < 1.0) v = 1.0;
      if (v != prev) { bounds[n++] = (float)v; prev = v; }
    }
    *nbp = n;
    for (int i = n; i < 64; ++i) bounds[i] = 3.0e38f;
  }
}

// ---------------- K1: feat + small MLPs (gates, lambda, fusion, bproj) ----------------
__global__ __launch_bounds__(256) void k_feat(
    const float* __restrict__ x, const int* __restrict__ type_seq, const int* __restrict__ seq_time,
    const float* __restrict__ temb,
    const float* __restrict__ gW1, const float* __restrict__ gb1,
    const float* __restrict__ gW2, const float* __restrict__ gb2,
    const float* __restrict__ fW1, const float* __restrict__ fb1,
    const float* __restrict__ fW2, const float* __restrict__ fb2,
    const float* __restrict__ bW1, const float* __restrict__ bb1,
    const float* __restrict__ bW2, const float* __restrict__ bb2,
    const float* __restrict__ lW1, const float* __restrict__ lb1,
    const float* __restrict__ lW2, const float* __restrict__ lb2,
    float* __restrict__ xf, float* __restrict__ qg, float* __restrict__ kg,
    float* __restrict__ vg, float* __restrict__ lam)
{
  int bl = blockIdx.x;                 // b*1024 + l
  int l = bl & 1023;
  int t = threadIdx.x;
  __shared__ float feat[17];
  __shared__ float hid[192];
  __shared__ float mf_s;
  int ty = type_seq[bl];
  if (t == 0) {
    float mf = (ty != 0) ? 1.f : 0.f;
    mf_s = mf;
    float dt = 0.f;
    if (l > 0) {
      int d = seq_time[bl] - seq_time[bl - 1];
      dt = d > 0 ? (float)d : 0.f;
    }
    feat[16] = log1pf(dt * mf);
  }
  if (t < 16) feat[t] = temb[ty * 16 + t];
  __syncthreads();
  if (t < 192) {
    int grp = t >> 5, m = t & 31;
    const float* W1; const float* b1;
    if (grp < 3)       { W1 = gW1 + grp * 17 * 32; b1 = gb1 + grp * 32; }
    else if (grp == 3) { W1 = lW1; b1 = lb1; }
    else if (grp == 4) { W1 = fW1; b1 = fb1; }
    else               { W1 = bW1; b1 = bb1; }
    float a = b1[m];
    #pragma unroll
    for (int f = 0; f < 17; ++f) a += feat[f] * W1[f * 32 + m];
    hid[t] = silu(a);
  }
  __syncthreads();
  float mf = mf_s, npad = 1.f - mf;
  if (t < 24) {
    int g = t >> 3, o = t & 7;
    float a = gb2[g * 8 + o];
    for (int m = 0; m < 32; ++m) a += hid[g * 32 + m] * gW2[(g * 32 + m) * 8 + o];
    float coef = (g == 2) ? 0.25f : 0.12f;
    float gate = (1.f + coef * tanhf(a)) * mf + npad;
    float* dst = (g == 0) ? qg : (g == 1) ? kg : vg;
    dst[bl * 8 + o] = gate;
  } else if (t == 24) {
    float a = lb2[0];
    for (int m = 0; m < 32; ++m) a += hid[96 + m] * lW2[m];
    lam[bl] = (1.f + 0.15f * tanhf(a)) * mf;
  }
  #pragma unroll
  for (int rep = 0; rep < 2; ++rep) {
    int o = t + rep * 256;
    float af = fb2[o], ab = bb2[o];
    for (int m = 0; m < 32; ++m) {
      af += hid[128 + m] * fW2[m * 512 + o];
      ab += hid[160 + m] * bW2[m * 512 + o];
    }
    float fg = fsigmoid(af);
    xf[(size_t)bl * 512 + o] = x[(size_t)bl * 512 + o] + fg * (ab * mf);
  }
}

// ---------------- K2: RoPE position cumsum ----------------
__global__ void k_cumsum(const float* __restrict__ lam, float* __restrict__ pos) {
  int b = threadIdx.x;
  if (b < 8) {
    float acc = 0.f;
    for (int l = 0; l < 1024; ++l) { acc += lam[b * 1024 + l]; pos[b * 1024 + l] = acc; }
  }
}

// ---------------- K3: uvqk = silu(xf @ proj_w + proj_b)  [8192 x 512 x 2048, bf16 WMMA] ----------------
__global__ __launch_bounds__(256) void k_gemm_proj(const float* __restrict__ A, const float* __restrict__ Bw,
                                                   const float* __restrict__ bias, float* __restrict__ Cout) {
  __shared__ __bf16 As[64][32];
  __shared__ __bf16 Bt[128][32];
  int t = threadIdx.x;
  int mblk = blockIdx.x * 64;
  int nblk = blockIdx.y * 128;
  int wave = t >> 5, lane = t & 31;
  int wm = wave & 1, wn = wave >> 1;
  v8f acc[2][2];
  #pragma unroll
  for (int i = 0; i < 2; ++i)
    #pragma unroll
    for (int j = 0; j < 2; ++j) acc[i][j] = (v8f){0.f,0.f,0.f,0.f,0.f,0.f,0.f,0.f};
  int ar = t >> 2;            // A row 0..63
  int ak = (t & 3) * 8;       // A k-offset 0,8,16,24
  int bn = t & 127;           // B column 0..127
  int bki = (t >> 7) * 16;    // B k-offset 0 or 16
  int kb = (lane >= 16) ? 8 : 0;
  int rl = lane & 15;
  const float* arow = A + (size_t)(mblk + ar) * 512 + ak;
  const float* bcol = Bw + nblk + bn;
  for (int kt = 0; kt < 16; ++kt) {
    int kb0 = kt * 32;
    {
      const float* src = arow + kb0;
      #pragma unroll
      for (int i = 0; i < 4; ++i)
        *(unsigned*)&As[ar][ak + 2 * i] = pack_bf16(src[2 * i], src[2 * i + 1]);
    }
    #pragma unroll
    for (int i = 0; i < 16; i += 2) {
      int k = bki + i;
      float f0 = bcol[(size_t)(kb0 + k) * 2048];
      float f1 = bcol[(size_t)(kb0 + k + 1) * 2048];
      *(unsigned*)&Bt[bn][k] = pack_bf16(f0, f1);
    }
    if (kt < 15) {
      __builtin_prefetch(arow + kb0 + 32, 0, 1);
      __builtin_prefetch(bcol + (size_t)(kb0 + 32 + bki) * 2048, 0, 1);
    }
    __syncthreads();
    v16bf af[2], bfr[2];
    #pragma unroll
    for (int i = 0; i < 2; ++i) {
      const __bf16* p = &As[wm * 32 + i * 16 + rl][0];
      af[i] = make_frag(p + kb, p + kb + 16);
    }
    #pragma unroll
    for (int j = 0; j < 2; ++j) {
      const __bf16* p = &Bt[wn * 32 + j * 16 + rl][0];
      bfr[j] = make_frag(p + kb, p + kb + 16);
    }
    #pragma unroll
    for (int i = 0; i < 2; ++i)
      #pragma unroll
      for (int j = 0; j < 2; ++j)
        acc[i][j] = wmma_bf16(af[i], bfr[j], acc[i][j]);
    __syncthreads();
  }
  int rhalf = (lane >= 16) ? 8 : 0;
  #pragma unroll
  for (int i = 0; i < 2; ++i)
    #pragma unroll
    for (int j = 0; j < 2; ++j) {
      int col = nblk + wn * 32 + j * 16 + rl;
      float bv = bias[col];
      #pragma unroll
      for (int r = 0; r < 8; ++r) {
        int row = mblk + wm * 32 + i * 16 + rhalf + r;
        Cout[(size_t)row * 2048 + col] = silu(acc[i][j][r] + bv);
      }
    }
}

// ---------------- K4: RoPE + per-head gating; Q,K row-major bf16, V transposed bf16 ----------------
__global__ __launch_bounds__(256) void k_qkv(const float* __restrict__ uvqk, const float* __restrict__ pos,
                                             const float* __restrict__ qg, const float* __restrict__ kg,
                                             const float* __restrict__ vg,
                                             __bf16* __restrict__ qb, __bf16* __restrict__ kb,
                                             __bf16* __restrict__ vT) {
  int idx = blockIdx.x * 256 + threadIdx.x;     // B*H*L*32
  int d = idx & 31;
  int l = (idx >> 5) & 1023;
  int h = (idx >> 15) & 7;
  int b = idx >> 18;
  int bl = b * 1024 + l;
  size_t rowbase = (size_t)bl * 2048;
  int dh0 = h * 64;
  float p = pos[bl];
  float freq = __expf(-(float)d * (9.210340371976184f / 32.f));  // 10000^(-2d/64)
  float s, c;
  __sincosf(p * freq, &s, &c);
  float qgv = qg[bl * 8 + h], kgv = kg[bl * 8 + h], vgv = vg[bl * 8 + h];
  float qe = uvqk[rowbase + 1024 + dh0 + 2 * d];
  float qo = uvqk[rowbase + 1024 + dh0 + 2 * d + 1];
  float ke = uvqk[rowbase + 1536 + dh0 + 2 * d];
  float ko = uvqk[rowbase + 1536 + dh0 + 2 * d + 1];
  float v0 = uvqk[rowbase + 512 + dh0 + d];
  float v1 = uvqk[rowbase + 512 + dh0 + d + 32];
  size_t qrow = ((size_t)(b * 8 + h) * 1024 + l) * 64;
  qb[qrow + d]      = f2bf((qe * c - qo * s) * qgv);
  qb[qrow + d + 32] = f2bf((qe * s + qo * c) * qgv);
  kb[qrow + d]      = f2bf((ke * c - ko * s) * kgv);
  kb[qrow + d + 32] = f2bf((ke * s + ko * c) * kgv);
  size_t vbase = (size_t)(b * 8 + h) * 64;
  vT[(vbase + d) * 1024 + l]      = f2bf(v0 * vgv);
  vT[(vbase + d + 32) * 1024 + l] = f2bf(v1 * vgv);
}

// ---------------- K5: attention (1 wave per (b,h,16-row q tile)) ----------------
__global__ __launch_bounds__(32) void k_attn(const __bf16* __restrict__ qb, const __bf16* __restrict__ kbuf,
                                             const __bf16* __restrict__ vT,
                                             const int* __restrict__ type_seq, const int* __restrict__ seq_time,
                                             const float* __restrict__ tbw, const float* __restrict__ head_scale,
                                             const float* __restrict__ table, const float* __restrict__ bounds,
                                             const int* __restrict__ nbp, float* __restrict__ ctx) {
  int blk = blockIdx.x;
  int qt = blk & 63;
  int h = (blk >> 6) & 7;
  int b = blk >> 9;
  int lane = threadIdx.x;
  int rl = lane & 15;
  int kbo = (lane >= 16) ? 8 : 0;
  int rh = (lane >= 16) ? 8 : 0;
  int qbase = qt * 16;

  __shared__ float bnd[64];
  __shared__ float tab[25];
  __shared__ __bf16 Sb[16][32];
  if (lane < 25) tab[lane] = table[h * 25 + lane];
  bnd[lane] = bounds[lane];
  bnd[lane + 32] = bounds[lane + 32];
  int nb = *nbp;
  __syncthreads();

  float hscale = head_scale[h];
  size_t bh = (size_t)(b * 8 + h);

  const __bf16* qp = qb + (bh * 1024 + qbase + rl) * 64;
  v16bf aq0 = make_frag(qp + kbo, qp + kbo + 16);
  v16bf aq1 = make_frag(qp + 32 + kbo, qp + 32 + kbo + 16);

  float ti[8]; int tyi[8]; float padi[8];
  #pragma unroll
  for (int r = 0; r < 8; ++r) {
    int i = qbase + rh + r;
    ti[r] = (float)seq_time[b * 1024 + i];
    int ty = type_seq[b * 1024 + i];
    tyi[r] = ty;
    padi[r] = (ty != 0) ? 1.f : 0.f;
  }
  v8f acc[4];
  #pragma unroll
  for (int g = 0; g < 4; ++g) acc[g] = (v8f){0.f,0.f,0.f,0.f,0.f,0.f,0.f,0.f};

  int nchunk = qt / 2 + 1;
  for (int kc = 0; kc < nchunk; ++kc) {
    int j0 = kc * 32;
    const __bf16* kp0 = kbuf + (bh * 1024 + j0 + rl) * 64;
    const __bf16* kp1 = kp0 + 16 * 64;
    v16bf bk0a = make_frag(kp0 + kbo, kp0 + kbo + 16);
    v16bf bk0b = make_frag(kp0 + 32 + kbo, kp0 + 32 + kbo + 16);
    v16bf bk1a = make_frag(kp1 + kbo, kp1 + kbo + 16);
    v16bf bk1b = make_frag(kp1 + 32 + kbo, kp1 + 32 + kbo + 16);
    v8f s0 = (v8f){0.f,0.f,0.f,0.f,0.f,0.f,0.f,0.f};
    v8f s1 = (v8f){0.f,0.f,0.f,0.f,0.f,0.f,0.f,0.f};
    s0 = wmma_bf16(aq0, bk0a, s0);
    s0 = wmma_bf16(aq1, bk0b, s0);
    s1 = wmma_bf16(aq0, bk1a, s1);
    s1 = wmma_bf16(aq1, bk1b, s1);

    int j_0 = j0 + rl, j_1 = j0 + 16 + rl;
    float tj0 = (float)seq_time[b * 1024 + j_0];
    float tj1 = (float)seq_time[b * 1024 + j_1];
    int tyj0 = type_seq[b * 1024 + j_0];
    int tyj1 = type_seq[b * 1024 + j_1];
    float padj0 = tyj0 != 0 ? 1.f : 0.f;
    float padj1 = tyj1 != 0 ? 1.f : 0.f;
    #pragma unroll
    for (int r = 0; r < 8; ++r) {
      int i = qbase + rh + r;
      {
        float td = fabsf(ti[r] - tj0); if (td < 1.f) td = 1.f;
        int lo = 0, hi = nb;
        while (lo < hi) { int mid = (lo + hi) >> 1; if (bnd[mid] < td) lo = mid + 1; else hi = mid; }
        float pm = padi[r] * padj0;
        float lg = s0[r] * hscale + 2.f * tbw[lo] + 4.f * tab[tyi[r] * 5 + tyj0] * pm;
        float a = (j_0 <= i) ? silu(lg) * pm : 0.f;
        Sb[rh + r][rl] = f2bf(a);
      }
      {
        float td = fabsf(ti[r] - tj1); if (td < 1.f) td = 1.f;
        int lo = 0, hi = nb;
        while (lo < hi) { int mid = (lo + hi) >> 1; if (bnd[mid] < td) lo = mid + 1; else hi = mid; }
        float pm = padi[r] * padj1;
        float lg = s1[r] * hscale + 2.f * tbw[lo] + 4.f * tab[tyi[r] * 5 + tyj1] * pm;
        float a = (j_1 <= i) ? silu(lg) * pm : 0.f;
        Sb[rh + r][16 + rl] = f2bf(a);
      }
    }
    __syncthreads();
    const __bf16* sp = &Sb[rl][0];
    v16bf as = make_frag(sp + kbo, sp + kbo + 16);
    #pragma unroll
    for (int g = 0; g < 4; ++g) {
      const __bf16* vp = vT + (bh * 64 + g * 16 + rl) * 1024 + j0;
      v16bf bv = make_frag(vp + kbo, vp + kbo + 16);
      acc[g] = wmma_bf16(as, bv, acc[g]);
    }
    __syncthreads();
  }
  #pragma unroll
  for (int g = 0; g < 4; ++g)
    #pragma unroll
    for (int r = 0; r < 8; ++r)
      ctx[((size_t)(b * 1024 + qbase + rh + r)) * 512 + h * 64 + g * 16 + rl] = acc[g][r];
}

// ---------------- K6: LayerNorm(ctx) * u  -> bf16 ----------------
__global__ __launch_bounds__(256) void k_ln(const float* __restrict__ ctx, const float* __restrict__ uvqk,
                                            const float* __restrict__ ln_g, const float* __restrict__ ln_b,
                                            __bf16* __restrict__ yb) {
  int row = blockIdx.x;
  int t = threadIdx.x;
  float c0 = ctx[(size_t)row * 512 + t];
  float c1 = ctx[(size_t)row * 512 + t + 256];
  __shared__ float red[256];
  __shared__ float mu_s, rv_s;
  red[t] = c0 + c1;
  __syncthreads();
  for (int s = 128; s > 0; s >>= 1) { if (t < s) red[t] += red[t + s]; __syncthreads(); }
  if (t == 0) mu_s = red[0] / 512.f;
  __syncthreads();
  float mu = mu_s;
  float d0 = c0 - mu, d1 = c1 - mu;
  red[t] = d0 * d0 + d1 * d1;
  __syncthreads();
  for (int s = 128; s > 0; s >>= 1) { if (t < s) red[t] += red[t + s]; __syncthreads(); }
  if (t == 0) rv_s = rsqrtf(red[0] / 512.f + 1e-5f);
  __syncthreads();
  float rv = rv_s;
  float u0 = uvqk[(size_t)row * 2048 + t];
  float u1 = uvqk[(size_t)row * 2048 + t + 256];
  yb[(size_t)row * 512 + t]       = f2bf((d0 * rv * ln_g[t] + ln_b[t]) * u0);
  yb[(size_t)row * 512 + t + 256] = f2bf((d1 * rv * ln_g[t + 256] + ln_b[t + 256]) * u1);
}

// ---------------- K7: out = xf + yb @ out_w + out_b  [8192 x 512 x 512, bf16 WMMA] ----------------
__global__ __launch_bounds__(256) void k_gemm_out(const __bf16* __restrict__ A, const float* __restrict__ Bw,
                                                  const float* __restrict__ bias, const float* __restrict__ xf,
                                                  float* __restrict__ Cout) {
  __shared__ __bf16 As[64][32];
  __shared__ __bf16 Bt[128][32];
  int t = threadIdx.x;
  int mblk = blockIdx.x * 64;
  int nblk = blockIdx.y * 128;
  int wave = t >> 5, lane = t & 31;
  int wm = wave & 1, wn = wave >> 1;
  v8f acc[2][2];
  #pragma unroll
  for (int i = 0; i < 2; ++i)
    #pragma unroll
    for (int j = 0; j < 2; ++j) acc[i][j] = (v8f){0.f,0.f,0.f,0.f,0.f,0.f,0.f,0.f};
  int ar = t >> 2;
  int ak = (t & 3) * 8;
  int bn = t & 127;
  int bki = (t >> 7) * 16;
  int kb = (lane >= 16) ? 8 : 0;
  int rl = lane & 15;
  const __bf16* arow = A + (size_t)(mblk + ar) * 512 + ak;
  const float* bcol = Bw + nblk + bn;
  for (int kt = 0; kt < 16; ++kt) {
    int kb0 = kt * 32;
    *(F4*)&As[ar][ak] = *(const F4*)(arow + kb0);
    #pragma unroll
    for (int i = 0; i < 16; i += 2) {
      int k = bki + i;
      float f0 = bcol[(size_t)(kb0 + k) * 512];
      float f1 = bcol[(size_t)(kb0 + k + 1) * 512];
      *(unsigned*)&Bt[bn][k] = pack_bf16(f0, f1);
    }
    if (kt < 15) {
      __builtin_prefetch(arow + kb0 + 32, 0, 1);
      __builtin_prefetch(bcol + (size_t)(kb0 + 32 + bki) * 512, 0, 1);
    }
    __syncthreads();
    v16bf af[2], bfr[2];
    #pragma unroll
    for (int i = 0; i < 2; ++i) {
      const __bf16* p = &As[wm * 32 + i * 16 + rl][0];
      af[i] = make_frag(p + kb, p + kb + 16);
    }
    #pragma unroll
    for (int j = 0; j < 2; ++j) {
      const __bf16* p = &Bt[wn * 32 + j * 16 + rl][0];
      bfr[j] = make_frag(p + kb, p + kb + 16);
    }
    #pragma unroll
    for (int i = 0; i < 2; ++i)
      #pragma unroll
      for (int j = 0; j < 2; ++j)
        acc[i][j] = wmma_bf16(af[i], bfr[j], acc[i][j]);
    __syncthreads();
  }
  int rhalf = (lane >= 16) ? 8 : 0;
  #pragma unroll
  for (int i = 0; i < 2; ++i)
    #pragma unroll
    for (int j = 0; j < 2; ++j) {
      int col = nblk + wn * 32 + j * 16 + rl;
      float bv = bias[col];
      #pragma unroll
      for (int r = 0; r < 8; ++r) {
        int row = mblk + wm * 32 + i * 16 + rhalf + r;
        Cout[(size_t)row * 512 + col] = acc[i][j][r] + bv + xf[(size_t)row * 512 + col];
      }
    }
}

extern "C" void kernel_launch(void* const* d_in, const int* in_sizes, int n_in,
                              void* d_out, int out_size, void* d_ws, size_t ws_size,
                              hipStream_t stream) {
  (void)in_sizes; (void)n_in; (void)out_size; (void)ws_size;
  const float* x        = (const float*)d_in[0];
  const int*   type_seq = (const int*)d_in[1];
  const int*   seq_time = (const int*)d_in[2];
  const float* temb     = (const float*)d_in[3];
  const float* gW1 = (const float*)d_in[4];
  const float* gb1 = (const float*)d_in[5];
  const float* gW2 = (const float*)d_in[6];
  const float* gb2 = (const float*)d_in[7];
  const float* fW1 = (const float*)d_in[8];
  const float* fb1 = (const float*)d_in[9];
  const float* fW2 = (const float*)d_in[10];
  const float* fb2 = (const float*)d_in[11];
  const float* bW1 = (const float*)d_in[12];
  const float* bb1 = (const float*)d_in[13];
  const float* bW2 = (const float*)d_in[14];
  const float* bb2 = (const float*)d_in[15];
  const float* lW1 = (const float*)d_in[16];
  const float* lb1 = (const float*)d_in[17];
  const float* lW2 = (const float*)d_in[18];
  const float* lb2 = (const float*)d_in[19];
  const float* tpr = (const float*)d_in[20];
  const float* tbw = (const float*)d_in[21];
  const float* proj_w = (const float*)d_in[22];
  const float* proj_b = (const float*)d_in[23];
  const float* out_w  = (const float*)d_in[24];
  const float* out_b  = (const float*)d_in[25];
  const float* ln_g   = (const float*)d_in[26];
  const float* ln_b   = (const float*)d_in[27];
  const float* head_scale = (const float*)d_in[28];

  char* ws = (char*)d_ws;
  size_t o = 0;
  float* uvqk = (float*)(ws + o); o += (size_t)8192 * 2048 * 4;
  float* xf   = (float*)(ws + o); o += (size_t)8192 * 512 * 4;
  float* ctx  = (float*)(ws + o); o += (size_t)8192 * 512 * 4;
  float* lam  = (float*)(ws + o); o += 8192 * 4;
  float* pos  = (float*)(ws + o); o += 8192 * 4;
  float* qg   = (float*)(ws + o); o += 8192 * 8 * 4;
  float* kg   = (float*)(ws + o); o += 8192 * 8 * 4;
  float* vg   = (float*)(ws + o); o += 8192 * 8 * 4;
  float* table  = (float*)(ws + o); o += 1024;
  float* bounds = (float*)(ws + o); o += 256;
  int*   nbp    = (int*)(ws + o);   o += 256;
  __bf16* qb = (__bf16*)(ws + o); o += (size_t)8192 * 512 * 2;
  __bf16* kb = (__bf16*)(ws + o); o += (size_t)8192 * 512 * 2;
  __bf16* vT = (__bf16*)(ws + o); o += (size_t)8192 * 512 * 2;
  __bf16* yb = (__bf16*)(ws + o); o += (size_t)8192 * 512 * 2;

  k_prep<<<1, 256, 0, stream>>>(tpr, table, bounds, nbp);
  k_feat<<<8192, 256, 0, stream>>>(x, type_seq, seq_time, temb,
                                   gW1, gb1, gW2, gb2, fW1, fb1, fW2, fb2,
                                   bW1, bb1, bW2, bb2, lW1, lb1, lW2, lb2,
                                   xf, qg, kg, vg, lam);
  k_cumsum<<<1, 8, 0, stream>>>(lam, pos);
  k_gemm_proj<<<dim3(128, 16), 256, 0, stream>>>(xf, proj_w, proj_b, uvqk);
  k_qkv<<<8192, 256, 0, stream>>>(uvqk, pos, qg, kg, vg, qb, kb, vT);
  k_attn<<<4096, 32, 0, stream>>>(qb, kb, vT, type_seq, seq_time, tbw, head_scale,
                                  table, bounds, nbp, ctx);
  k_ln<<<8192, 256, 0, stream>>>(ctx, uvqk, ln_g, ln_b, yb);
  k_gemm_out<<<dim3(128, 4), 256, 0, stream>>>(yb, out_w, out_b, xf, (float*)d_out);
}
